// DenseALIGNN_27066883899808
// MI455X (gfx1250) — matrix operand order repaired and checked by reading, not compiled, not run on hardware
//
#include <hip/hip_runtime.h>
#include <hip/hip_bf16.h>

// ---------------------------------------------------------------------------
// Types for CDNA5 WMMA (wave32, v_wmma_f32_16x16x32_bf16)
// ---------------------------------------------------------------------------
typedef __bf16 bf16_t;
typedef __attribute__((ext_vector_type(16))) __bf16 v16bf;
typedef __attribute__((ext_vector_type(8)))  __bf16 v8bf;
typedef __attribute__((ext_vector_type(8)))  float  v8f;

#define GEMM_WAVES 8   // 256 threads = 8 wave32 per block

// ---------------------------------------------------------------------------
// C[M,N] = A[M,Kpad] @ Wt[Npad,Kpad]^T (+bias)(+resid); bf16 WMMA, f32 accum.
// A: bf16, lda >= Kpad, pad columns finite (multiplied by zero weights).
// Wt: bf16, transposed + zero-padded (row n = column n of W, Kpad contiguous).
// One wave computes a 16-row x (NT*16)-col strip: A fragment loaded once,
// reused for NT WMMAs. All fragment loads are unguarded 16B vector loads.
// ---------------------------------------------------------------------------
template <int NT>
__global__ __launch_bounds__(256) void gemm_wmma_k(
    const bf16_t* __restrict__ A, int lda,
    const bf16_t* __restrict__ Wt, int ldwt,   // ldwt == Kpad
    const float* __restrict__ bias,
    const float* __restrict__ resid, int ldr,
    float* __restrict__ C, int ldc,
    int M, int N, int Kpad)
{
    const int wave = threadIdx.x >> 5;
    const int lane = threadIdx.x & 31;
    const int row0 = (blockIdx.y * GEMM_WAVES + wave) * 16;
    if (row0 >= M) return;                 // uniform per wave: EXEC stays all-1s
    const int r = lane & 15;
    const int h = lane >> 4;
    int arow = row0 + r;
    if (arow >= M) arow = M - 1;           // clamp: garbage rows never stored
    const bf16_t* __restrict__ ap = A  + (size_t)arow * lda;
    const bf16_t* __restrict__ bp = Wt + (size_t)r * ldwt;

    v8f acc[NT];
#pragma unroll
    for (int t = 0; t < NT; ++t) acc[t] = (v8f){};

    for (int k0 = 0; k0 < Kpad; k0 += 32) {
        // A fragment (ISA 7.12.2 16-bit A): elems 0..7 -> K=k0+h*8+j,
        // elems 8..15 -> K=k0+16+h*8+j  => two contiguous 16B loads
        v8bf alo = *reinterpret_cast<const v8bf*>(ap + k0 + h * 8);
        v8bf ahi = *reinterpret_cast<const v8bf*>(ap + k0 + 16 + h * 8);
        v16bf a = __builtin_shufflevector(alo, ahi,
                    0,1,2,3,4,5,6,7,8,9,10,11,12,13,14,15);
#pragma unroll
        for (int t = 0; t < NT; ++t) {
            // B fragment: lane holds col (t*16+r); elems 0..7 -> K=k0+h*8+j, elems 8..15 -> K=k0+16+h*8+j
            const bf16_t* bpt = bp + (size_t)t * 16 * ldwt + k0 + h * 8;
            v8bf blo = *reinterpret_cast<const v8bf*>(bpt);
            v8bf bhi = *reinterpret_cast<const v8bf*>(bpt + 16);
            v16bf b = __builtin_shufflevector(blo, bhi,
                        0,1,2,3,4,5,6,7,8,9,10,11,12,13,14,15);
            acc[t] = __builtin_amdgcn_wmma_f32_16x16x32_bf16(
                false, a, false, b, (short)0, acc[t], false, false);
        }
    }
    // C/D layout: VGPR v, lanes0-15 -> M=v ; lanes16-31 -> M=v+8 ; N = lane&15
#pragma unroll
    for (int t = 0; t < NT; ++t) {
#pragma unroll
        for (int v = 0; v < 8; ++v) {
            int rr = row0 + v + 8 * h;
            int cc = t * 16 + r;
            if (rr < M && cc < N) {
                float val = acc[t][v];
                if (bias)  val += bias[cc];
                if (resid) val += resid[(size_t)rr * ldr + cc];
                C[(size_t)rr * ldc + cc] = val;
            }
        }
    }
}

// ---------------------------------------------------------------------------
// Weight pack: W[K,N] f32 -> Wt[Npad,Kpad] bf16, transposed, zero-padded.
// ---------------------------------------------------------------------------
__global__ __launch_bounds__(256) void pack_w_k(
    const float* __restrict__ W, bf16_t* __restrict__ Wt,
    int K, int N, int Kpad, int Npad)
{
    int idx = blockIdx.x * blockDim.x + threadIdx.x;
    if (idx >= Kpad * Npad) return;
    int n = idx / Kpad;
    int k = idx % Kpad;
    Wt[idx] = (k < K && n < N) ? (bf16_t)W[(size_t)k * N + n] : (bf16_t)0.0f;
}

// Activation pack: X[M,K] f32 -> Xp[M,Kpad] bf16 zero-padded.
__global__ __launch_bounds__(256) void pack_a_k(
    const float* __restrict__ X, bf16_t* __restrict__ Xp, int M, int K, int Kpad)
{
    size_t idx = (size_t)blockIdx.x * blockDim.x + threadIdx.x;
    if (idx >= (size_t)M * Kpad) return;
    int row = (int)(idx / Kpad);
    int k   = (int)(idx % Kpad);
    Xp[idx] = (k < K) ? (bf16_t)X[(size_t)row * K + k] : (bf16_t)0.0f;
}

// ---------------------------------------------------------------------------
// Row-wise LayerNorm + SiLU: Y = silu((X - mean)*rsqrt(var+1e-5)*g + b)
// ---------------------------------------------------------------------------
template <typename IT, typename OT>
__global__ __launch_bounds__(256) void ln_silu_k(
    const IT* __restrict__ X, int ldx,
    const float* __restrict__ g, const float* __restrict__ b,
    OT* __restrict__ Y, int ldy, int M, int D)
{
    int row = blockIdx.x * 8 + (threadIdx.x >> 5);
    if (row >= M) return;
    int lane = threadIdx.x & 31;
    const IT* xr = X + (size_t)row * ldx;
    float s = 0.0f, ss = 0.0f;
    for (int j = lane; j < D; j += 32) {
        float v = (float)xr[j];
        s += v; ss += v * v;
    }
#pragma unroll
    for (int o = 16; o > 0; o >>= 1) {
        s  += __shfl_xor(s,  o, 32);
        ss += __shfl_xor(ss, o, 32);
    }
    float inv_d = 1.0f / (float)D;
    float mean  = s * inv_d;
    float var   = ss * inv_d - mean * mean;
    float inv   = rsqrtf(var + 1e-5f);
    OT* yr = Y + (size_t)row * ldy;
    for (int j = lane; j < D; j += 32) {
        float v  = ((float)xr[j] - mean) * inv * g[j] + b[j];
        float sg = 1.0f / (1.0f + __expf(-v));
        yr[j] = (OT)(v * sg);
    }
}

// ---------------------------------------------------------------------------
// RBF expansion (bf16 out): out[i,j] = exp(-gamma*(d[i] - (vmin+j*step))^2)
// ---------------------------------------------------------------------------
__global__ __launch_bounds__(256) void rbf_k(
    const float* __restrict__ d, bf16_t* __restrict__ out,
    int n, int bins, int ldo, float vmin, float step, float gamma)
{
    size_t idx = (size_t)blockIdx.x * blockDim.x + threadIdx.x;
    if (idx >= (size_t)n * bins) return;
    int i = (int)(idx / bins);
    int j = (int)(idx % bins);
    float c = vmin + step * (float)j;
    float df = d[i] - c;
    out[(size_t)i * ldo + j] = (bf16_t)__expf(-gamma * df * df);
}

// ---------------------------------------------------------------------------
// Edge gate + scatter: e = Xsg[src]+Xdg[dst]+Ye ; s = sigmoid(e)
//   e_out[e] = e ; num[dst] += Xdu[src]*s ; den[dst] += s   (64 gate dims)
// ---------------------------------------------------------------------------
template <typename ET>
__global__ __launch_bounds__(256) void egc_scatter_k(
    const float* __restrict__ xg_s, const float* __restrict__ xg_d,
    const float* __restrict__ xu_d, const float* __restrict__ ye,
    const int* __restrict__ src, const int* __restrict__ dst,
    ET* __restrict__ e_out, int lde,
    float* __restrict__ num, float* __restrict__ den, int E)
{
    size_t idx = (size_t)blockIdx.x * blockDim.x + threadIdx.x;
    if (idx >= (size_t)E * 64) return;
    int e = (int)(idx >> 6);
    int d = (int)(idx & 63);
    int sr = src[e], dr = dst[e];
    float val = xg_s[(size_t)sr * 64 + d] + xg_d[(size_t)dr * 64 + d]
              + ye[(size_t)e * 64 + d];
    e_out[(size_t)e * lde + d] = (ET)val;
    float sg = 1.0f / (1.0f + __expf(-val));
    atomicAdd(&num[(size_t)dr * 64 + d], xu_d[(size_t)sr * 64 + d] * sg);
    atomicAdd(&den[(size_t)dr * 64 + d], sg);
}

// out[n] = Xsu[n] + num[n] / (den[n] + 1e-6)
__global__ __launch_bounds__(256) void egc_finalize_k(
    const float* __restrict__ xu_s, const float* __restrict__ num,
    const float* __restrict__ den, float* __restrict__ out, int ldo, int Nn)
{
    size_t idx = (size_t)blockIdx.x * blockDim.x + threadIdx.x;
    if (idx >= (size_t)Nn * 64) return;
    int n = (int)(idx >> 6);
    int d = (int)(idx & 63);
    out[(size_t)n * ldo + d] =
        xu_s[(size_t)n * 64 + d] + num[(size_t)n * 64 + d] / (den[(size_t)n * 64 + d] + 1e-6f);
}

// Graph mean-pool accumulation
__global__ __launch_bounds__(256) void pool_k(
    const float* __restrict__ x, int ldx, const int* __restrict__ gid,
    float* __restrict__ hsum, float* __restrict__ cnt, int Nn)
{
    size_t idx = (size_t)blockIdx.x * blockDim.x + threadIdx.x;
    if (idx >= (size_t)Nn * 92) return;
    int n = (int)(idx / 92);
    int d = (int)(idx % 92);
    int g = gid[n];
    atomicAdd(&hsum[(size_t)g * 92 + d], x[(size_t)n * ldx + d]);
    if (d == 0) atomicAdd(&cnt[g], 1.0f);
}

// out[g] = (hsum[g] . W) / max(cnt,1) + b    (one wave per graph)
__global__ __launch_bounds__(32) void head_k(
    const float* __restrict__ hsum, const float* __restrict__ cnt,
    const float* __restrict__ Wf, const float* __restrict__ bf,
    float* __restrict__ out)
{
    int g = blockIdx.x;
    int lane = threadIdx.x;
    float s = 0.0f;
    for (int j = lane; j < 92; j += 32) s += hsum[(size_t)g * 92 + j] * Wf[j];
#pragma unroll
    for (int o = 16; o > 0; o >>= 1) s += __shfl_xor(s, o, 32);
    if (lane == 0) out[g] = s / fmaxf(cnt[g], 1.0f) + bf[0];
}

// ---------------------------------------------------------------------------
// Host orchestration
// ---------------------------------------------------------------------------
#define N_NODES   10000
#define N_EDGES   100000
#define N_TRIP    600000
#define N_GRAPHS  64
#define XLD 288
#define YLD 288
#define ZLD 224

extern "C" void kernel_launch(void* const* d_in, const int* in_sizes, int n_in,
                              void* d_out, int out_size, void* d_ws, size_t ws_size,
                              hipStream_t stream)
{
    (void)in_sizes; (void)out_size;
    if (n_in < 123) return;   // 8 data inputs + 115 param leaves expected

    const float* atomF  = (const float*)d_in[0];
    const float* bond   = (const float*)d_in[1];
    const float* angle  = (const float*)d_in[2];
    const int*   g_src  = (const int*)d_in[3];
    const int*   g_dst  = (const int*)d_in[4];
    const int*   lg_src = (const int*)d_in[5];
    const int*   lg_dst = (const int*)d_in[6];
    const int*   gid    = (const int*)d_in[7];
    auto PP = [&](int i) -> const float* { return (const float*)d_in[8 + i]; };

    // Param leaf indices (JAX pytree: dict keys sorted lexicographically)
    const int ABX = 0, ABY = 3, ALN = 6, ANG1 = 60, ANG2 = 64, ATOM = 68,
              EDG1 = 72, EDG2 = 76, FC = 80, GBX = 82, GCN0 = 88;

    // ---- bump allocator over d_ws ----
    size_t off = 0;
    auto alloc = [&](size_t elems, size_t esz) -> void* {
        off = (off + 255) & ~(size_t)255;
        void* p = (char*)d_ws + off;
        off += elems * esz;
        return p;
    };
    float*  xcat   = (float*) alloc((size_t)N_NODES * XLD, 4);
    float*  ycat   = (float*) alloc((size_t)N_EDGES * YLD, 4);
    bf16_t* zcat   = (bf16_t*)alloc((size_t)N_TRIP  * ZLD, 2);
    bf16_t* bigA   = (bf16_t*)alloc((size_t)N_TRIP  * ZLD, 2);
    float*  bigB   = (float*) alloc((size_t)N_TRIP  * 96,  4);
    bf16_t* nodeT1 = (bf16_t*)alloc((size_t)N_EDGES * YLD, 2);
    bf16_t* nodeT2 = (bf16_t*)alloc((size_t)N_NODES * XLD, 2);
    bf16_t* atomP  = (bf16_t*)alloc((size_t)N_NODES * 96,  2);
    float*  eY     = (float*) alloc((size_t)N_EDGES * 64,  4);
    bf16_t* eYln   = (bf16_t*)alloc((size_t)N_EDGES * 64,  2);
    float*  g1     = (float*) alloc((size_t)N_EDGES * 64,  4);
    float*  g2     = (float*) alloc((size_t)N_EDGES * 64,  4);
    float*  g3     = (float*) alloc((size_t)N_EDGES * 64,  4);
    float*  g4     = (float*) alloc((size_t)N_EDGES * 64,  4);
    float*  num    = (float*) alloc((size_t)N_EDGES * 64,  4);
    float*  den    = (float*) alloc((size_t)N_EDGES * 64,  4);
    float*  xfin   = (float*) alloc((size_t)N_NODES * 96,  4);
    float*  hsum   = (float*) alloc((size_t)N_GRAPHS * 92, 4);
    float*  cnt    = (float*) alloc((size_t)N_GRAPHS,      4);
    bf16_t* wpool  = (bf16_t*)alloc((size_t)1536 * 1024,   2);  // packed weights
    if (off > ws_size) return;

    // ---- helpers ----
    size_t woff = 0;
    auto gemm = [&](const bf16_t* A, int lda, const float* Wf, int K,
                    const float* bias, const float* resid, int ldr,
                    float* C, int ldc, int M, int N) {
        int Kpad = (K + 31) & ~31;
        int Npad = (N + 15) & ~15;
        int NT   = Npad / 16;          // 4 (N=64) or 6 (N=92)
        bf16_t* Wt = wpool + woff;
        woff += (size_t)Kpad * Npad;
        int tot = Kpad * Npad;
        pack_w_k<<<(tot + 255) / 256, 256, 0, stream>>>(Wf, Wt, K, N, Kpad, Npad);
        dim3 g(1, (M + 16 * GEMM_WAVES - 1) / (16 * GEMM_WAVES));
        if (NT == 4)
            gemm_wmma_k<4><<<g, 256, 0, stream>>>(A, lda, Wt, Kpad, bias, resid, ldr, C, ldc, M, N, Kpad);
        else
            gemm_wmma_k<6><<<g, 256, 0, stream>>>(A, lda, Wt, Kpad, bias, resid, ldr, C, ldc, M, N, Kpad);
    };
    auto lnFB = [&](const float* X, int ldx, const float* g, const float* b,
                    bf16_t* Y, int ldy, int M, int D) {
        ln_silu_k<float, bf16_t><<<(M + 7) / 8, 256, 0, stream>>>(X, ldx, g, b, Y, ldy, M, D);
    };
    auto lnFF = [&](const float* X, int ldx, const float* g, const float* b,
                    float* Y, int ldy, int M, int D) {
        ln_silu_k<float, float><<<(M + 7) / 8, 256, 0, stream>>>(X, ldx, g, b, Y, ldy, M, D);
    };
    auto lnBB = [&](const bf16_t* X, int ldx, const float* g, const float* b,
                    bf16_t* Y, int ldy, int M, int D) {
        ln_silu_k<bf16_t, bf16_t><<<(M + 7) / 8, 256, 0, stream>>>(X, ldx, g, b, Y, ldy, M, D);
    };
    auto grid1 = [](size_t n) { return (unsigned)((n + 255) / 256); };

    // ================= Embeddings =================
    // atom: silu(ln(atomF @ W + b)) -> xcat[:,0:92]
    pack_a_k<<<grid1((size_t)N_NODES * 96), 256, 0, stream>>>(atomF, atomP, N_NODES, 92, 96);
    gemm(atomP, 96, PP(ATOM + 0), 92, PP(ATOM + 1), nullptr, 0, bigB, 96, N_NODES, 92);
    lnFF(bigB, 96, PP(ATOM + 3), PP(ATOM + 2), xcat, XLD, N_NODES, 92);

    // edge: rbf(bond; 0..8, 81 bins, ls=0.5 -> gamma=4) -> mlp -> mlp -> ycat[:,0:92]
    rbf_k<<<grid1((size_t)N_EDGES * 81), 256, 0, stream>>>(
        bond, bigA, N_EDGES, 81, 96, 0.0f, 8.0f / 80.0f, 4.0f);
    gemm(bigA, 96, PP(EDG1 + 0), 81, PP(EDG1 + 1), nullptr, 0, bigB, 96, N_EDGES, 92);
    lnFB(bigB, 96, PP(EDG1 + 3), PP(EDG1 + 2), nodeT1, 96, N_EDGES, 92);
    gemm(nodeT1, 96, PP(EDG2 + 0), 92, PP(EDG2 + 1), nullptr, 0, bigB, 96, N_EDGES, 92);
    lnFF(bigB, 96, PP(EDG2 + 3), PP(EDG2 + 2), ycat, YLD, N_EDGES, 92);

    // angle: rbf(angle; -pi..pi, 40 bins -> gamma=39/2pi) -> mlp -> mlp -> zcat[:,0:92]
    const float PI = 3.14159265358979323846f;
    rbf_k<<<grid1((size_t)N_TRIP * 40), 256, 0, stream>>>(
        angle, bigA, N_TRIP, 40, 64, -PI, 2.0f * PI / 39.0f, 39.0f / (2.0f * PI));
    gemm(bigA, 64, PP(ANG1 + 0), 40, PP(ANG1 + 1), nullptr, 0, bigB, 96, N_TRIP, 92);
    lnFB(bigB, 96, PP(ANG1 + 3), PP(ANG1 + 2), bigA, 96, N_TRIP, 92);
    gemm(bigA, 96, PP(ANG2 + 0), 92, PP(ANG2 + 1), nullptr, 0, bigB, 96, N_TRIP, 92);
    lnFB(bigB, 96, PP(ANG2 + 3), PP(ANG2 + 2), zcat, ZLD, N_TRIP, 92);

    // ================= DenseALIGNN block =================
    for (int i = 0; i < 3; ++i) {
        const int fin = 92 + 64 * i;
        const int eb_ = ALN + i * 18;       // edge egc leaves
        const int nb_ = ALN + i * 18 + 9;   // node egc leaves
        // ---- node egc on graph (nodes <- edges) ----
        lnFB(xcat, XLD, PP(nb_ + 8), PP(nb_ + 7), nodeT2, XLD, N_NODES, fin);
        lnFB(ycat, YLD, PP(nb_ + 6), PP(nb_ + 5), nodeT1, YLD, N_EDGES, fin);
        gemm(nodeT2, XLD, PP(nb_ + 3), fin, nullptr, nullptr, 0, g1, 64, N_NODES, 64);   // Wsg
        gemm(nodeT2, XLD, PP(nb_ + 0), fin, nullptr, nullptr, 0, g2, 64, N_NODES, 64);   // Wdg
        gemm(nodeT2, XLD, PP(nb_ + 4), fin, nullptr, nullptr, 0, g3, 64, N_NODES, 64);   // Wsu
        gemm(nodeT2, XLD, PP(nb_ + 1), fin, nullptr, nullptr, 0, g4, 64, N_NODES, 64);   // Wdu
        gemm(nodeT1, YLD, PP(nb_ + 2), fin, nullptr, nullptr, 0, bigB, 64, N_EDGES, 64); // Weg
        hipMemsetAsync(num, 0, (size_t)N_NODES * 64 * 4, stream);
        hipMemsetAsync(den, 0, (size_t)N_NODES * 64 * 4, stream);
        egc_scatter_k<float><<<grid1((size_t)N_EDGES * 64), 256, 0, stream>>>(
            g1, g2, g4, bigB, g_src, g_dst, eY, 64, num, den, N_EDGES);
        egc_finalize_k<<<grid1((size_t)N_NODES * 64), 256, 0, stream>>>(
            g3, num, den, xcat + 92 + 64 * i, XLD, N_NODES);
        // ---- edge egc on line graph (edges <- triplets), nfeat = eY ----
        lnFB(eY, 64, PP(eb_ + 8), PP(eb_ + 7), eYln, 64, N_EDGES, 64);
        lnBB(zcat, ZLD, PP(eb_ + 6), PP(eb_ + 5), bigA, ZLD, N_TRIP, fin);
        gemm(eYln, 64, PP(eb_ + 3), 64, nullptr, nullptr, 0, g1, 64, N_EDGES, 64);
        gemm(eYln, 64, PP(eb_ + 0), 64, nullptr, nullptr, 0, g2, 64, N_EDGES, 64);
        gemm(eYln, 64, PP(eb_ + 4), 64, nullptr, nullptr, 0, g3, 64, N_EDGES, 64);
        gemm(eYln, 64, PP(eb_ + 1), 64, nullptr, nullptr, 0, g4, 64, N_EDGES, 64);
        gemm(bigA, ZLD, PP(eb_ + 2), fin, nullptr, nullptr, 0, bigB, 64, N_TRIP, 64);
        hipMemsetAsync(num, 0, (size_t)N_EDGES * 64 * 4, stream);
        hipMemsetAsync(den, 0, (size_t)N_EDGES * 64 * 4, stream);
        egc_scatter_k<bf16_t><<<grid1((size_t)N_TRIP * 64), 256, 0, stream>>>(
            g1, g2, g4, bigB, lg_src, lg_dst, zcat + 92 + 64 * i, ZLD, num, den, N_TRIP);
        egc_finalize_k<<<grid1((size_t)N_EDGES * 64), 256, 0, stream>>>(
            g3, num, den, ycat + 92 + 64 * i, YLD, N_EDGES);
    }

    // ---- bottlenecks: x = x0 + silu(ln(xcat)) @ abx.W  (same for y / aby) ----
    lnFB(xcat, XLD, PP(ABX + 2), PP(ABX + 1), nodeT2, XLD, N_NODES, 284);
    gemm(nodeT2, XLD, PP(ABX + 0), 284, nullptr, xcat, XLD, xcat, XLD, N_NODES, 92);
    lnFB(ycat, YLD, PP(ABY + 2), PP(ABY + 1), nodeT1, YLD, N_EDGES, 284);
    gemm(nodeT1, YLD, PP(ABY + 0), 284, nullptr, ycat, YLD, ycat, YLD, N_EDGES, 92);

    // ================= DenseGCN block =================
    for (int i = 0; i < 3; ++i) {
        const int fin = 92 + 64 * i;
        const int pb = GCN0 + i * 9;
        lnFB(xcat, XLD, PP(pb + 8), PP(pb + 7), nodeT2, XLD, N_NODES, fin);
        lnFB(ycat, YLD, PP(pb + 6), PP(pb + 5), nodeT1, YLD, N_EDGES, fin);
        gemm(nodeT2, XLD, PP(pb + 3), fin, nullptr, nullptr, 0, g1, 64, N_NODES, 64);
        gemm(nodeT2, XLD, PP(pb + 0), fin, nullptr, nullptr, 0, g2, 64, N_NODES, 64);
        gemm(nodeT2, XLD, PP(pb + 4), fin, nullptr, nullptr, 0, g3, 64, N_NODES, 64);
        gemm(nodeT2, XLD, PP(pb + 1), fin, nullptr, nullptr, 0, g4, 64, N_NODES, 64);
        gemm(nodeT1, YLD, PP(pb + 2), fin, nullptr, nullptr, 0, bigB, 64, N_EDGES, 64);
        hipMemsetAsync(num, 0, (size_t)N_NODES * 64 * 4, stream);
        hipMemsetAsync(den, 0, (size_t)N_NODES * 64 * 4, stream);
        egc_scatter_k<float><<<grid1((size_t)N_EDGES * 64), 256, 0, stream>>>(
            g1, g2, g4, bigB, g_src, g_dst, ycat + 92 + 64 * i, YLD, num, den, N_EDGES);
        egc_finalize_k<<<grid1((size_t)N_NODES * 64), 256, 0, stream>>>(
            g3, num, den, xcat + 92 + 64 * i, XLD, N_NODES);
    }

    // gbx bottleneck -> xfin
    lnFB(xcat, XLD, PP(GBX + 2), PP(GBX + 1), nodeT2, XLD, N_NODES, 284);
    gemm(nodeT2, XLD, PP(GBX + 0), 284, nullptr, xcat, XLD, xfin, 96, N_NODES, 92);

    // ================= Readout =================
    hipMemsetAsync(hsum, 0, (size_t)N_GRAPHS * 92 * 4, stream);
    hipMemsetAsync(cnt, 0, (size_t)N_GRAPHS * 4, stream);
    pool_k<<<grid1((size_t)N_NODES * 92), 256, 0, stream>>>(xfin, 96, gid, hsum, cnt, N_NODES);
    head_k<<<N_GRAPHS, 32, 0, stream>>>(hsum, cnt, PP(FC + 0), PP(FC + 1), (float*)d_out);
}
